// GCNResnet_84688165143176
// MI455X (gfx1250) — matrix operand order repaired and checked by reading, not compiled, not run on hardware
//
#include <hip/hip_runtime.h>

// ---------- types ----------
typedef __attribute__((ext_vector_type(16))) __bf16 v16bf;
typedef __attribute__((ext_vector_type(8)))  float  v8f;
typedef __attribute__((ext_vector_type(4)))  unsigned int v4u;

union FragBF { v4u u4[2]; v16bf b; };
union Pack8  { unsigned short s[8]; v4u v; };

#define LSTR 56   // LDS row stride (ushorts): 112B -> 16B aligned b128, conflict-free frag gathers
#define ASZ (128 * LSTR)
#define BSZ (64 * LSTR)

// ---------- helpers ----------
__device__ __forceinline__ unsigned short f2bf(float f) {
  unsigned int u = __float_as_uint(f);
  u += 0x7FFFu + ((u >> 16) & 1u);          // RNE
  return (unsigned short)(u >> 16);
}
__device__ __forceinline__ void atomicMaxF(float* a, float v) {
  if (v >= 0.0f) atomicMax((int*)a, __float_as_int(v));
  else           atomicMin((unsigned int*)a, __float_as_uint(v));
}

// =====================================================================
// bf16 WMMA GEMM: C = epilogue(A[M,K] @ B^T), B stored [N][K] (K-contiguous).
// Block tile 128(M) x 64(N), 8 waves each own 32x32 (4 wmma / K-chunk).
// Double-buffered LDS pipeline: one barrier per chunk; next chunk staged in
// registers while WMMAs drain. transC=1 stores C^T[n][m] (b128 along channels).
// pack16 remaps n(0..195) into a 16x16 interior. poolmax fuses residual+maxpool.
// =====================================================================
__launch_bounds__(256)
__global__ void wmma_gemm(const unsigned short* __restrict__ A,
                          const unsigned short* __restrict__ B,
                          float* __restrict__ Cf, unsigned short* __restrict__ Cb,
                          int M, int N, int K, int ldb, int ldc,
                          long long strideB, long long strideC,
                          const float* __restrict__ biasM, const float* __restrict__ scaleM,
                          const float* __restrict__ shiftM, const float* __restrict__ biasN,
                          int act, int transC, int pack16,
                          const float* __restrict__ fres, float* __restrict__ pooled, int poolmax)
{
  __shared__ __align__(16) unsigned short As[2 * ASZ];
  __shared__ __align__(16) unsigned short Bs[2 * BSZ];

  const int tid  = threadIdx.x;
  const int lane = tid & 31;
  const int wave = tid >> 5;
  const int g    = lane >> 4;        // K-half selector in frags
  const int nl   = lane & 15;
  const int mg   = wave & 3;         // wave's 32-row group
  const int ng   = wave >> 2;        // wave's 32-col group
  const int m0   = blockIdx.y * 128;
  const int n0   = blockIdx.x * 64;
  const int z    = blockIdx.z;
  const unsigned short* Bz = B + (long long)z * strideB;

  v8f acc[2][2];
#pragma unroll
  for (int i = 0; i < 2; ++i)
#pragma unroll
    for (int j = 0; j < 2; ++j) acc[i][j] = v8f{0.f,0.f,0.f,0.f,0.f,0.f,0.f,0.f};

  const int arow = tid >> 1, ac16 = (tid & 1) * 16;      // A: 128 rows x 32k
  const int brow = tid >> 2, bc8  = (tid & 3) * 8;       // B: 64 rows x 32k
  const unsigned short* pA = A  + (long long)(m0 + arow) * K + ac16;
  const unsigned short* pB = Bz + (long long)(n0 + brow) * ldb + bc8;

  auto do_mma = [&](const unsigned short* Ab, const unsigned short* Bb) {
    FragBF fa[2], fb[2];
#pragma unroll
    for (int i = 0; i < 2; ++i) {
      const unsigned short* pa = &Ab[(mg * 32 + i * 16 + nl) * LSTR];
      fa[i].u4[0] = *(const v4u*)(pa + g * 8);
      fa[i].u4[1] = *(const v4u*)(pa + 16 + g * 8);
    }
#pragma unroll
    for (int j = 0; j < 2; ++j) {
      const unsigned short* pb = &Bb[(ng * 32 + j * 16 + nl) * LSTR + g * 8];
      fb[j].u4[0] = *(const v4u*)(pb);
      fb[j].u4[1] = *(const v4u*)(pb + 16);
    }
#pragma unroll
    for (int i = 0; i < 2; ++i)
#pragma unroll
      for (int j = 0; j < 2; ++j)
        acc[i][j] = __builtin_amdgcn_wmma_f32_16x16x32_bf16(
            false, fa[i].b, false, fb[j].b, (short)0, acc[i][j], false, false);
  };
  auto store_tiles = [&](int buf, v4u a0, v4u a1, v4u b0) {
    *(v4u*)(&As[buf * ASZ + arow * LSTR + ac16])     = a0;
    *(v4u*)(&As[buf * ASZ + arow * LSTR + ac16 + 8]) = a1;
    *(v4u*)(&Bs[buf * BSZ + brow * LSTR + bc8])      = b0;
  };
  auto load_slow = [&](int kk) {
    long long gm = m0 + arow;
#pragma unroll
    for (int h = 0; h < 2; ++h) {
      int kb = kk + ac16 + h * 8;
#pragma unroll
      for (int i = 0; i < 8; ++i)
        As[arow * LSTR + ac16 + h * 8 + i] =
            (gm < (long long)M && kb + i < K) ? A[gm * (long long)K + kb + i] : (unsigned short)0;
    }
    int gn = n0 + brow, kb = kk + bc8;
#pragma unroll
    for (int i = 0; i < 8; ++i)
      Bs[brow * LSTR + bc8 + i] =
          (gn < N && kb + i < K) ? Bz[(long long)gn * ldb + kb + i] : (unsigned short)0;
  };

  const bool fast = (m0 + 128 <= M) && (n0 + 64 <= N);
  const int Kfull = fast ? (K & ~31) : 0;
  const int nch = Kfull >> 5;

  if (nch > 0) {                                    // pipelined main loop
    store_tiles(0, *(const v4u*)pA, *(const v4u*)(pA + 8), *(const v4u*)pB);
    __syncthreads();
    for (int c = 0; c < nch - 1; ++c) {
      const int kk = (c + 1) << 5;
      if (kk + 32 < Kfull) {                        // near-scope prefetch, 2 chunks ahead
        __builtin_prefetch(pA + kk + 32, 0, 3);
        __builtin_prefetch(pB + kk + 32, 0, 3);
      }
      v4u a0 = *(const v4u*)(pA + kk);
      v4u a1 = *(const v4u*)(pA + kk + 8);
      v4u b0 = *(const v4u*)(pB + kk);
      do_mma(As + (c & 1) * ASZ, Bs + (c & 1) * BSZ);
      store_tiles((c + 1) & 1, a0, a1, b0);
      __syncthreads();
    }
    do_mma(As + ((nch - 1) & 1) * ASZ, Bs + ((nch - 1) & 1) * BSZ);
    __syncthreads();
  }
  for (int kk = Kfull; kk < K; kk += 32) {          // masked edge/tail chunks (buffer 0)
    load_slow(kk);
    __syncthreads();
    do_mma(As, Bs);
    __syncthreads();
  }

  // ---- epilogue
  float* Cfz = Cf ? Cf + (long long)z * strideC : (float*)0;
  unsigned short* Cbz = Cb ? Cb + (long long)z * strideC : (unsigned short*)0;
#pragma unroll
  for (int i = 0; i < 2; ++i) {
#pragma unroll
    for (int j = 0; j < 2; ++j) {
      int n = n0 + ng * 32 + j * 16 + nl;
      int mbase = m0 + mg * 32 + i * 16 + g * 8;
      if (poolmax) {
        if (n < N) {
#pragma unroll
          for (int r = 0; r < 8; ++r) {
            int m = mbase + r;
            float v = acc[i][j][r] + biasM[m];
            float f = fres[(long long)z * (long long)M * N + (long long)m * N + n];
            atomicMaxF(&pooled[(long long)z * M + m], 0.5f * f + 0.5f * v);
          }
        }
        continue;
      }
      if (transC) {
        if (n >= N) continue;
        int row = pack16 ? (n + 2 * (n / 14) + 17) : n;
        if (Cbz && mbase + 8 <= M) {                // vector path: 8 consecutive channels
          Pack8 o;
#pragma unroll
          for (int r = 0; r < 8; ++r) {
            int m = mbase + r;
            float v = acc[i][j][r];
            if (biasM)  v += biasM[m];
            if (scaleM) v = v * scaleM[m] + shiftM[m];
            if (act == 1)      v = v > 0.f ? v : 0.f;
            else if (act == 2) v = 1.f / (1.f + __expf(-v));
            else if (act == 3) v = v > 0.f ? v : 0.2f * v;
            o.s[r] = f2bf(v);
          }
          *(v4u*)(&Cbz[(long long)row * ldc + mbase]) = o.v;
        } else {
#pragma unroll
          for (int r = 0; r < 8; ++r) {
            int m = mbase + r;
            if (m >= M) continue;
            float v = acc[i][j][r];
            if (biasM)  v += biasM[m];
            if (scaleM) v = v * scaleM[m] + shiftM[m];
            if (biasN)  v += biasN[n];
            if (act == 1)      v = v > 0.f ? v : 0.f;
            else if (act == 2) v = 1.f / (1.f + __expf(-v));
            else if (act == 3) v = v > 0.f ? v : 0.2f * v;
            long long o = (long long)row * ldc + m;
            if (Cbz) Cbz[o] = f2bf(v); else Cfz[o] = v;
          }
        }
      } else {                                      // row-major store (small GEMMs)
        if (n >= N) continue;
#pragma unroll
        for (int r = 0; r < 8; ++r) {
          int m = mbase + r;
          if (m >= M) continue;
          float v = acc[i][j][r];
          if (biasM)  v += biasM[m];
          if (scaleM) v = v * scaleM[m] + shiftM[m];
          if (biasN)  v += biasN[n];
          if (act == 1)      v = v > 0.f ? v : 0.f;
          else if (act == 2) v = 1.f / (1.f + __expf(-v));
          else if (act == 3) v = v > 0.f ? v : 0.2f * v;
          long long o = (long long)m * ldc + n;
          if (Cbz) Cbz[o] = f2bf(v); else Cfz[o] = v;
        }
      }
    }
  }
}

// =====================================================================
// 3x3 valid conv (16x16 -> 14x14) as 9 shifted K=1024 GEMMs, flattened into
// a single 288-chunk double-buffered pipeline (no flush at tap boundaries).
// Wr: bf16 [9][1024][1024]; In: [z][256][1024] (pixel-major); Out: [z][196][1024].
// =====================================================================
__launch_bounds__(256)
__global__ void wmma_conv3(const unsigned short* __restrict__ Wr,
                           const unsigned short* __restrict__ In,
                           unsigned short* __restrict__ Out,
                           const float* __restrict__ biasM, const float* __restrict__ scaleM,
                           const float* __restrict__ shiftM, int relu)
{
  __shared__ __align__(16) unsigned short As[2 * ASZ];
  __shared__ __align__(16) unsigned short Bs[2 * BSZ];

  const int tid  = threadIdx.x;
  const int lane = tid & 31, wave = tid >> 5;
  const int g = lane >> 4, nl = lane & 15;
  const int mg = wave & 3, ng = wave >> 2;
  const int m0 = blockIdx.y * 128, n0 = blockIdx.x * 64;
  const long long z = blockIdx.z;
  const unsigned short* Inz = In + z * (256ll * 1024);

  v8f acc[2][2];
#pragma unroll
  for (int i = 0; i < 2; ++i)
#pragma unroll
    for (int j = 0; j < 2; ++j) acc[i][j] = v8f{0.f,0.f,0.f,0.f,0.f,0.f,0.f,0.f};

  const int arow = tid >> 1, ac16 = (tid & 1) * 16;
  const int brow = tid >> 2, bc8  = (tid & 3) * 8;
  const int gnb = n0 + brow;
  const int by = gnb / 14, bx = gnb - 14 * by;
  const bool bvalid = gnb < 196;

  auto addrA = [&](int c) {
    int t = c >> 5, kk = (c & 31) << 5;
    return Wr + (long long)t * (1024 * 1024) + (long long)(m0 + arow) * 1024 + ac16 + kk;
  };
  auto addrB = [&](int c) {
    int t = c >> 5, kk = (c & 31) << 5;
    int dy = t / 3, dx = t - 3 * dy;
    int p = (by + dy) * 16 + (bx + dx);
    return Inz + (long long)p * 1024 + bc8 + kk;
  };
  auto do_mma = [&](const unsigned short* Ab, const unsigned short* Bb) {
    FragBF fa[2], fb[2];
#pragma unroll
    for (int i = 0; i < 2; ++i) {
      const unsigned short* pa = &Ab[(mg * 32 + i * 16 + nl) * LSTR];
      fa[i].u4[0] = *(const v4u*)(pa + g * 8);
      fa[i].u4[1] = *(const v4u*)(pa + 16 + g * 8);
    }
#pragma unroll
    for (int j = 0; j < 2; ++j) {
      const unsigned short* pb = &Bb[(ng * 32 + j * 16 + nl) * LSTR + g * 8];
      fb[j].u4[0] = *(const v4u*)(pb);
      fb[j].u4[1] = *(const v4u*)(pb + 16);
    }
#pragma unroll
    for (int i = 0; i < 2; ++i)
#pragma unroll
      for (int j = 0; j < 2; ++j)
        acc[i][j] = __builtin_amdgcn_wmma_f32_16x16x32_bf16(
            false, fa[i].b, false, fb[j].b, (short)0, acc[i][j], false, false);
  };
  auto store_tiles = [&](int buf, v4u a0, v4u a1, v4u b0) {
    *(v4u*)(&As[buf * ASZ + arow * LSTR + ac16])     = a0;
    *(v4u*)(&As[buf * ASZ + arow * LSTR + ac16 + 8]) = a1;
    *(v4u*)(&Bs[buf * BSZ + brow * LSTR + bc8])      = b0;
  };

  const int NCH = 288;                               // 9 taps x 32 K-chunks
  {
    const unsigned short* a = addrA(0);
    const unsigned short* b = addrB(0);
    store_tiles(0, *(const v4u*)a, *(const v4u*)(a + 8),
                bvalid ? *(const v4u*)b : v4u{0u,0u,0u,0u});
  }
  __syncthreads();
  for (int c = 0; c < NCH - 1; ++c) {
    const unsigned short* a = addrA(c + 1);
    const unsigned short* b = addrB(c + 1);
    if (c + 2 < NCH) {                               // near-scope prefetch, 2 chunks ahead
      __builtin_prefetch(addrA(c + 2), 0, 3);
      __builtin_prefetch(addrB(c + 2), 0, 3);
    }
    v4u a0 = *(const v4u*)a;
    v4u a1 = *(const v4u*)(a + 8);
    v4u b0 = bvalid ? *(const v4u*)b : v4u{0u,0u,0u,0u};
    do_mma(As + (c & 1) * ASZ, Bs + (c & 1) * BSZ);
    store_tiles((c + 1) & 1, a0, a1, b0);
    __syncthreads();
  }
  do_mma(As + ((NCH - 1) & 1) * ASZ, Bs + ((NCH - 1) & 1) * BSZ);

  unsigned short* Oz = Out + z * (196ll * 1024);
#pragma unroll
  for (int i = 0; i < 2; ++i) {
#pragma unroll
    for (int j = 0; j < 2; ++j) {
      int n = n0 + ng * 32 + j * 16 + nl;
      if (n >= 196) continue;
      int mbase = m0 + mg * 32 + i * 16 + g * 8;
      Pack8 o;
#pragma unroll
      for (int r = 0; r < 8; ++r) {
        int m = mbase + r;
        float v = acc[i][j][r] + biasM[m];
        if (scaleM) v = v * scaleM[m] + shiftM[m];
        if (relu)   v = v > 0.f ? v : 0.f;
        o.s[r] = f2bf(v);
      }
      *(v4u*)(&Oz[(long long)n * 1024 + mbase]) = o.v;
    }
  }
}

// ---------- small utility kernels ----------
__global__ void cvt_f32_bf16_k(const float* __restrict__ in, unsigned short* __restrict__ out, long long n) {
  long long i = (long long)blockIdx.x * blockDim.x + threadIdx.x;
  if (i < n) out[i] = f2bf(in[i]);
}
// LDS-tiled transpose + convert: in f32 [z][R][C] -> out bf16 [z][C][R]
__global__ void cvtT_k(const float* __restrict__ in, unsigned short* __restrict__ out, int R, int C) {
  __shared__ float tile[32][33];
  const int c0 = blockIdx.x * 32, r0 = blockIdx.y * 32;
  const long long zb = blockIdx.z;
  const float* inz = in + zb * (long long)R * C;
  unsigned short* outz = out + zb * (long long)R * C;
  const int tx = threadIdx.x, ty = threadIdx.y;            // 32 x 8
#pragma unroll
  for (int j = 0; j < 4; ++j) {
    int r = r0 + ty + j * 8;
    if (r < R && c0 + tx < C) tile[ty + j * 8][tx] = inz[(long long)r * C + c0 + tx];
  }
  __syncthreads();
#pragma unroll
  for (int j = 0; j < 4; ++j) {
    int c = c0 + ty + j * 8;
    if (c < C && r0 + tx < R) outz[(long long)c * R + r0 + tx] = f2bf(tile[tx][ty + j * 8]);
  }
}
// OIHW [1024,1024,3,3] f32 -> bf16 [t][co][ci]
__global__ void repack3x3_k(const float* __restrict__ in, unsigned short* __restrict__ out) {
  long long i = (long long)blockIdx.x * blockDim.x + threadIdx.x;
  if (i >= 9ll * 1024 * 1024) return;
  int t = (int)(i / (1024 * 1024));
  long long rem = i - (long long)t * (1024 * 1024);
  out[i] = f2bf(in[rem * 9 + t]);
}
// fill [64][256][1024] with per-channel border value (interior overwritten later)
__global__ void fill16_k(unsigned short* __restrict__ out, const float* __restrict__ biasM,
                         const float* __restrict__ scaleM, const float* __restrict__ shiftM,
                         int relu) {
  long long i = (long long)blockIdx.x * blockDim.x + threadIdx.x;
  if (i >= 64ll * 256 * 1024) return;
  int c = (int)(i & 1023);
  float v = biasM[c];
  if (scaleM) v = v * scaleM[c] + shiftM[c];
  if (relu)   v = v > 0.f ? v : 0.f;
  out[i] = f2bf(v);
}
// softmax over 80 contiguous channels * gate -> bf16  (layout [b*196][80])
__global__ void softmax_mul_k(const float* __restrict__ att, const float* __restrict__ lin,
                              unsigned short* __restrict__ out) {
  int id = blockIdx.x * blockDim.x + threadIdx.x;
  if (id >= 64 * 196) return;
  long long base = (long long)id * 80;
  float mx = -1e30f;
  for (int c = 0; c < 80; ++c) mx = fmaxf(mx, att[base + c]);
  float sum = 0.f;
  for (int c = 0; c < 80; ++c) sum += __expf(att[base + c] - mx);
  float inv = 1.f / sum;
  for (int c = 0; c < 80; ++c)
    out[base + c] = f2bf(__expf(att[base + c] - mx) * inv * lin[base + c]);
}
__global__ void rowsum_rsqrt_k(const float* __restrict__ A, float* __restrict__ D) {
  int i = threadIdx.x;
  if (i < 80) { float s = 0.f; for (int j = 0; j < 80; ++j) s += A[i * 80 + j]; D[i] = rsqrtf(s); }
}
__global__ void adj_k(const float* __restrict__ A, const float* __restrict__ D,
                      unsigned short* __restrict__ adj) {
  int id = blockIdx.x * blockDim.x + threadIdx.x;
  if (id < 6400) { int i = id / 80, j = id - i * 80; adj[id] = f2bf(D[i] * A[j * 80 + i] * D[j]); }
}
__global__ void fill_neginf_k(float* __restrict__ p, int n) {
  int i = blockIdx.x * blockDim.x + threadIdx.x;
  if (i < n) p[i] = -3.0e38f;
}

// ---------- host-side helpers ----------
static void launch_gemm(hipStream_t st, const unsigned short* A, const unsigned short* B,
                        float* Cf, unsigned short* Cb, int M, int N, int K, int ldb, int ldc,
                        long long sB, long long sC, int batch,
                        const float* bM, const float* sM, const float* shM, const float* bN,
                        int act, int transC, int pack16,
                        const float* fres, float* pooled, int poolmax) {
  dim3 g((N + 63) / 64, (M + 127) / 128, batch);
  wmma_gemm<<<g, 256, 0, st>>>(A, B, Cf, Cb, M, N, K, ldb, ldc, sB, sC,
                               bM, sM, shM, bN, act, transC, pack16, fres, pooled, poolmax);
}

extern "C" void kernel_launch(void* const* d_in, const int* in_sizes, int n_in,
                              void* d_out, int out_size, void* d_ws, size_t ws_size,
                              hipStream_t stream)
{
  (void)in_sizes; (void)n_in; (void)out_size; (void)ws_size;
  const float* feature = (const float*)d_in[0];
  const float* inp     = (const float*)d_in[1];
  const float* A_adj   = (const float*)d_in[2];
  const float* A1w = (const float*)d_in[3],  *A1b = (const float*)d_in[4];
  const float* bn1s = (const float*)d_in[5], *bn1b = (const float*)d_in[6];
  const float* A2w = (const float*)d_in[7],  *A2b = (const float*)d_in[8];
  const float* bn2s = (const float*)d_in[9], *bn2b = (const float*)d_in[10];
  const float* A3w = (const float*)d_in[11], *A3b = (const float*)d_in[12];
  const float* bn3s = (const float*)d_in[13],*bn3b = (const float*)d_in[14];
  const float* L1w = (const float*)d_in[15], *L1b = (const float*)d_in[16];
  const float* lbn1s = (const float*)d_in[17],*lbn1b = (const float*)d_in[18];
  const float* L2w = (const float*)d_in[19], *L2b = (const float*)d_in[20];
  const float* K1w = (const float*)d_in[21], *K1b = (const float*)d_in[22];
  const float* K2w = (const float*)d_in[23], *K2b = (const float*)d_in[24];
  const float* K3w = (const float*)d_in[25], *K3b = (const float*)d_in[26];
  const float* gc1w = (const float*)d_in[27],*gc2w = (const float*)d_in[28];
  const float* Wimg = (const float*)d_in[29],*bimg = (const float*)d_in[30];
  const float* Wcls = (const float*)d_in[31],*bcls = (const float*)d_in[32];

  size_t off = 0;
  auto alloc = [&](size_t bytes) -> void* {
    void* p = (char*)d_ws + off;
    off += (bytes + 255) & ~(size_t)255;
    return p;
  };
  typedef unsigned short us;
  us* fb16T  = (us*)alloc(64ll*196*2048*2);   // [b][s][ci]
  us* att1T  = (us*)alloc(64ll*256*1024*2);   // [b][pix16][ci]
  us* att2T  = (us*)alloc(64ll*196*1024*2);
  float* att3T = (float*)alloc(64ll*196*80*4);
  us* lin1T  = (us*)alloc(64ll*196*1024*2);
  float* lin2T = (float*)alloc(64ll*196*80*4);
  us* attlin = (us*)alloc(64ll*196*80*2);
  us* link1T = (us*)alloc(64ll*256*1024*2);
  us* link2T = (us*)alloc(64ll*196*1024*2);
  us* wA1 = (us*)alloc(1024ll*2048*2);
  us* wA2r = (us*)alloc(9ll*1024*1024*2);
  us* wA3 = (us*)alloc(80ll*1024*2);
  us* wL1 = (us*)alloc(1024ll*2048*2);
  us* wL2 = (us*)alloc(80ll*1024*2);
  us* wK1 = (us*)alloc(1024ll*80*2);
  us* wK2r = (us*)alloc(9ll*1024*1024*2);
  us* wK3 = (us*)alloc(2048ll*1024*2);
  us* wgc1T = (us*)alloc(1024ll*300*2);       // [N][K]
  us* wgc2T = (us*)alloc(2048ll*1024*2);      // [N][K]
  us* wimg = (us*)alloc(1000ll*2048*2);       // already [N][K]
  us* wcls = (us*)alloc(1000ll*2048*2);
  us* inpb = (us*)alloc(80ll*300*2);
  us* adjb = (us*)alloc(80ll*80*2);
  float* Dv = (float*)alloc(80*4);
  float* pooled = (float*)alloc(64ll*2048*4);
  us* pooledb = (us*)alloc(64ll*2048*2);
  us* tmp1T = (us*)alloc(1024ll*80*2);        // [N][M]
  us* x1   = (us*)alloc(80ll*1024*2);         // row-major (A of next)
  us* tmp2T = (us*)alloc(2048ll*80*2);        // [N][M]
  us* x2   = (us*)alloc(80ll*2048*2);
  us* imgb = (us*)alloc(64ll*1000*2);
  us* clsb = (us*)alloc(80ll*1000*2);

  auto cvt = [&](const float* s, us* d, long long n) {
    cvt_f32_bf16_k<<<dim3((unsigned)((n + 255) / 256)), 256, 0, stream>>>(s, d, n);
  };

  // ---- conversions / repacks / transposes
  cvtT_k<<<dim3(7, 64, 64), dim3(32, 8), 0, stream>>>(feature, fb16T, 2048, 196);
  cvtT_k<<<dim3(32, 10, 1), dim3(32, 8), 0, stream>>>(gc1w, wgc1T, 300, 1024);
  cvtT_k<<<dim3(64, 32, 1), dim3(32, 8), 0, stream>>>(gc2w, wgc2T, 1024, 2048);
  cvt(A1w, wA1, 1024ll*2048);   cvt(A3w, wA3, 80ll*1024);
  cvt(L1w, wL1, 1024ll*2048);   cvt(L2w, wL2, 80ll*1024);
  cvt(K1w, wK1, 1024ll*80);     cvt(K3w, wK3, 2048ll*1024);
  cvt(Wimg, wimg, 1000ll*2048); cvt(Wcls, wcls, 1000ll*2048);
  cvt(inp, inpb, 80ll*300);
  repack3x3_k<<<dim3(36864), 256, 0, stream>>>(A2w, wA2r);
  repack3x3_k<<<dim3(36864), 256, 0, stream>>>(K2w, wK2r);
  rowsum_rsqrt_k<<<1, 128, 0, stream>>>(A_adj, Dv);
  adj_k<<<25, 256, 0, stream>>>(A_adj, Dv, adjb);

  // ---- attention branch
  fill16_k<<<dim3(65536), 256, 0, stream>>>(att1T, A1b, bn1s, bn1b, 1);
  launch_gemm(stream, wA1, fb16T, 0, att1T, 1024, 196, 2048, 2048, 1024,
              196ll*2048, 256ll*1024, 64, A1b, bn1s, bn1b, 0, 1, 1, 1, 0, 0, 0);
  wmma_conv3<<<dim3(4, 8, 64), 256, 0, stream>>>(wA2r, att1T, att2T, A2b, bn2s, bn2b, 1);
  launch_gemm(stream, wA3, att2T, att3T, 0, 80, 196, 1024, 1024, 80,
              196ll*1024, 196ll*80, 64, A3b, bn3s, bn3b, 0, 1, 1, 0, 0, 0, 0);

  // ---- linear branch
  launch_gemm(stream, wL1, fb16T, 0, lin1T, 1024, 196, 2048, 2048, 1024,
              196ll*2048, 196ll*1024, 64, L1b, lbn1s, lbn1b, 0, 1, 1, 0, 0, 0, 0);
  launch_gemm(stream, wL2, lin1T, lin2T, 0, 80, 196, 1024, 1024, 80,
              196ll*1024, 196ll*80, 64, L2b, 0, 0, 0, 2, 1, 0, 0, 0, 0);

  // ---- gate
  softmax_mul_k<<<dim3(49), 256, 0, stream>>>(att3T, lin2T, attlin);

  // ---- link branch
  fill16_k<<<dim3(65536), 256, 0, stream>>>(link1T, K1b, 0, 0, 0);
  launch_gemm(stream, wK1, attlin, 0, link1T, 1024, 196, 80, 80, 1024,
              196ll*80, 256ll*1024, 64, K1b, 0, 0, 0, 0, 1, 1, 0, 0, 0);
  wmma_conv3<<<dim3(4, 8, 64), 256, 0, stream>>>(wK2r, link1T, link2T, K2b, 0, 0, 0);
  fill_neginf_k<<<dim3(512), 256, 0, stream>>>(pooled, 64 * 2048);
  launch_gemm(stream, wK3, link2T, 0, 0, 2048, 196, 1024, 1024, 0,
              196ll*1024, 0, 64, K3b, 0, 0, 0, 0, 0, 0, feature, pooled, 1);
  cvt(pooled, pooledb, 64ll*2048);

  // ---- GCN
  launch_gemm(stream, inpb, wgc1T, 0, tmp1T, 80, 1024, 300, 300, 80, 0, 0, 1,
              0, 0, 0, 0, 0, 1, 0, 0, 0, 0);                   // (inp@gc1w)^T  [1024][80]
  launch_gemm(stream, adjb, tmp1T, 0, x1, 80, 1024, 80, 80, 1024, 0, 0, 1,
              0, 0, 0, 0, 3, 0, 0, 0, 0, 0);                   // leaky(adj@.) row-major
  launch_gemm(stream, x1, wgc2T, 0, tmp2T, 80, 2048, 1024, 1024, 80, 0, 0, 1,
              0, 0, 0, 0, 0, 1, 0, 0, 0, 0);                   // (x1@gc2w)^T [2048][80]
  launch_gemm(stream, adjb, tmp2T, 0, x2, 80, 2048, 80, 80, 2048, 0, 0, 1,
              0, 0, 0, 0, 0, 0, 0, 0, 0, 0);                   // adj@. row-major

  // ---- MFB head
  launch_gemm(stream, pooledb, wimg, 0, imgb, 64, 1000, 2048, 2048, 1000, 0, 0, 1,
              0, 0, 0, bimg, 0, 0, 0, 0, 0, 0);
  launch_gemm(stream, x2, wcls, 0, clsb, 80, 1000, 2048, 2048, 1000, 0, 0, 1,
              0, 0, 0, bcls, 0, 0, 0, 0, 0, 0);
  launch_gemm(stream, imgb, clsb, (float*)d_out, 0, 64, 80, 1000, 1000, 80, 0, 0, 1,
              0, 0, 0, 0, 0, 0, 0, 0, 0, 0);
}
